// EntmaxAttention_90142773608635
// MI455X (gfx1250) — matrix operand order repaired and checked by reading, not compiled, not run on hardware
//
#include <hip/hip_runtime.h>

// ---------------------------------------------------------------------------
// MI455X (gfx1250) entmax-1.5 multi-head attention, bf16 WMMA + fused entmax
// + TDM (tensor_load_to_lds) staging of K tiles.
//
// d_in: x, Wq, bq, Wk, bk, Wv, bv, Wo, bo (all f32)
// d_out: [ out f32 4*1024*1024 | attn f32 4*16*1024*1024 ]
// d_ws layout (bytes, needs >= 48 MiB):
//   xb   @ 0         : x as bf16              [4096][1024]
//   WqT  @ 8  MiB    : Wq^T bf16              [1024][1024]   (cout-major)
//   WkT  @ 10 MiB, WvT @ 12 MiB, WoT @ 14 MiB
//   qh   @ 16 MiB    : q bf16 head layout     [4][16][1024][64]
//   kh   @ 24 MiB, vh @ 32 MiB
//   oh   @ 40 MiB    : attn@v result bf16     [4096][1024]
// ---------------------------------------------------------------------------

typedef unsigned short ushort_t;
typedef __attribute__((ext_vector_type(8)))  float        v8f;
typedef __attribute__((ext_vector_type(8)))  __bf16       v8bf;
typedef __attribute__((ext_vector_type(16))) __bf16       v16bf;
typedef __attribute__((ext_vector_type(4)))  unsigned int u32x4;
typedef __attribute__((ext_vector_type(8)))  int          i32x8;
typedef __attribute__((ext_vector_type(4)))  int          i32x4;

struct alignas(16) B16cpy { unsigned long long a, b; };

__device__ inline ushort_t f2bf(float f) {
  unsigned int u = __float_as_uint(f);
  unsigned int r = (u + 0x7FFFu + ((u >> 16) & 1u)) >> 16;   // round-nearest-even
  return (ushort_t)r;
}

__device__ inline v16bf cat16(v8bf lo, v8bf hi) {
  return __builtin_shufflevector(lo, hi, 0,1,2,3,4,5,6,7,8,9,10,11,12,13,14,15);
}

__device__ inline v8f wmma_bf16(v16bf a, v16bf b, v8f c) {
  // D = A(16x32) * B(32x16) + C, f32 accumulate
  return __builtin_amdgcn_wmma_f32_16x16x32_bf16(false, a, false, b, (short)0, c,
                                                 false, false);
}

__device__ inline float wave_sum(float v) {
  #pragma unroll
  for (int m = 16; m; m >>= 1) v += __shfl_xor(v, m, 32);
  return v;
}
__device__ inline float wave_max(float v) {
  #pragma unroll
  for (int m = 16; m; m >>= 1) v = fmaxf(v, __shfl_xor(v, m, 32));
  return v;
}

// TDM: load a rows x cols tile of 2-byte elements (row stride = `stride`
// elements) from global `gaddr` into LDS byte-offset `ldsaddr`.
// D# per CDNA5 ISA (group0/group1; groups 2-3 unused).
// 6-arg builtin form (clang-23 / therock-10.0 toolchain).
__device__ inline void tdm_load_2d_b16(unsigned int ldsaddr,
                                       unsigned long long gaddr,
                                       int rows, int cols, int stride,
                                       int tensor_rows) {
  u32x4 g0;
  g0[0] = 1u;                                        // count=1, user mode
  g0[1] = ldsaddr;                                   // lds_addr (bytes)
  g0[2] = (unsigned int)(gaddr & 0xffffffffull);     // global_addr[31:0]
  g0[3] = (unsigned int)((gaddr >> 32) & 0x01ffffffull)
        | 0x80000000u;                               // addr[56:32] | type=2
  i32x8 g1;
  g1[0] = 0x00010000;                                // data_size=1 (2 bytes)
  g1[1] = (cols & 0xffff) << 16;                     // tensor_dim0[15:0]
  g1[2] = (tensor_rows & 0xffff) << 16;              // tensor_dim1[15:0]
  g1[3] = (cols & 0xffff) << 16;                     // tile_dim0
  g1[4] = rows & 0xffff;                             // tile_dim1
  g1[5] = stride;                                    // tensor_dim0_stride[31:0]
  g1[6] = 0;
  g1[7] = 0;
  i32x4 gz4 = {0, 0, 0, 0};
  i32x8 gz8 = {0, 0, 0, 0, 0, 0, 0, 0};
  __builtin_amdgcn_tensor_load_to_lds(g0, g1, gz4, gz4, gz8, 0);
}

// --------------------------- convert / transpose ---------------------------

__global__ __launch_bounds__(256) void cvt_bf16_kernel(const float* __restrict__ src,
                                                       ushort_t* __restrict__ dst, int n) {
  int i = blockIdx.x * 256 + threadIdx.x;
  if (i < n) dst[i] = f2bf(src[i]);
}

// W [1024][1024] f32  ->  WT [1024][1024] bf16 with WT[co][c] = W[c][co]
__global__ __launch_bounds__(256) void transpose_bf16_kernel(const float* __restrict__ W,
                                                             ushort_t* __restrict__ WT) {
  __shared__ ushort_t t[32][33];
  int bx = blockIdx.x * 32;            // co base
  int by = blockIdx.y * 32;            // c  base
  int tx = threadIdx.x & 31, ty = threadIdx.x >> 5;
  #pragma unroll
  for (int yy = ty; yy < 32; yy += 8)
    t[yy][tx] = f2bf(W[(size_t)(by + yy) * 1024 + bx + tx]);
  __syncthreads();
  #pragma unroll
  for (int yy = ty; yy < 32; yy += 8)
    WT[(size_t)(bx + yy) * 1024 + by + tx] = t[tx][yy];
}

// --------------------------------- GEMM ------------------------------------
// C[n][co] = sum_c A[n][c] * BT[co][c] + bias[co]
// A: [4096][1024] bf16 row-major, BT: [1024][1024] bf16 (cout-major).
// Block: 256 thr (8 waves), tile 128(M) x 128(N), K-step 32.
// Each wave: 32(M) x 64(N) -> acc[2][4], 8 WMMA per K-step.
// mode 0: write bf16 to head layout [B=4][H=16][N=1024][64]   (q/k/v)
// mode 1: write f32 to dst[n*1024+co]                          (final out)
__global__ __launch_bounds__(256) void gemm_bf16_kernel(const ushort_t* __restrict__ A,
                                                        const ushort_t* __restrict__ BT,
                                                        const float* __restrict__ bias,
                                                        void* __restrict__ dst, int mode) {
  __shared__ ushort_t sA[128 * 32];
  __shared__ ushort_t sB[128 * 32];
  int tid = threadIdx.x, wave = tid >> 5, lane = tid & 31;
  int gm0 = blockIdx.y * 128;
  int gn0 = blockIdx.x * 128;
  int wm = wave & 3, wn = wave >> 2;         // 4x2 wave grid, each 32(M)x64(N)

  v8f acc[2][4] = {};

  for (int k0 = 0; k0 < 1024; k0 += 32) {
    {   // stage A and B tiles, 128x32 bf16 each (two 16B copies per thread each)
      int e0 = tid * 16;
      const ushort_t* srcA = A  + (size_t)(gm0 + (e0 >> 5)) * 1024 + k0 + (e0 & 31);
      const ushort_t* srcB = BT + (size_t)(gn0 + (e0 >> 5)) * 1024 + k0 + (e0 & 31);
      *(B16cpy*)&sA[e0]     = *(const B16cpy*)srcA;
      *(B16cpy*)&sA[e0 + 8] = *(const B16cpy*)(srcA + 8);
      *(B16cpy*)&sB[e0]     = *(const B16cpy*)srcB;
      *(B16cpy*)&sB[e0 + 8] = *(const B16cpy*)(srcB + 8);
    }
    __syncthreads();

    int b0  = (lane & 16) ? 16 : 0;          // A K base (contiguous k = 16h + s)
    int b1  = (lane & 16) ? 24 : 8;          // A K base + 8
    int kb2 = (lane & 16) ? 16 : 0;          // B K base
    v16bf af[2], bf[4];
    #pragma unroll
    for (int i = 0; i < 2; ++i) {
      int row = wm * 32 + i * 16 + (lane & 15);
      af[i] = cat16(*(const v8bf*)&sA[row * 32 + b0],
                    *(const v8bf*)&sA[row * 32 + b1]);
    }
    #pragma unroll
    for (int j = 0; j < 4; ++j) {
      int col = wn * 64 + j * 16 + (lane & 15);
      bf[j] = cat16(*(const v8bf*)&sB[col * 32 + kb2],
                    *(const v8bf*)&sB[col * 32 + kb2 + 8]);
    }
    #pragma unroll
    for (int i = 0; i < 2; ++i)
      #pragma unroll
      for (int j = 0; j < 4; ++j)
        acc[i][j] = wmma_bf16(af[i], bf[j], acc[i][j]);
    __syncthreads();
  }

  // epilogue
  #pragma unroll
  for (int i = 0; i < 2; ++i)
    #pragma unroll
    for (int j = 0; j < 4; ++j)
      #pragma unroll
      for (int r = 0; r < 8; ++r) {
        int lrow = ((lane & 16) ? 8 : 0) + r;
        int grow = gm0 + wm * 32 + i * 16 + lrow;
        int gcol = gn0 + wn * 64 + j * 16 + (lane & 15);
        float val = acc[i][j][r] + bias[gcol];
        if (mode == 0) {
          int bb = grow >> 10, n = grow & 1023, hh = gcol >> 6, d = gcol & 63;
          ((ushort_t*)dst)[((((size_t)bb * 16 + hh) * 1024) + n) * 64 + d] = f2bf(val);
        } else {
          ((float*)dst)[(size_t)grow * 1024 + gcol] = val;
        }
      }
}

// --------------------------- fused attention -------------------------------
// One block per (b*h, 32-query-row tile). 256 threads = 8 waves.
// Dynamic LDS: sS 32x1024 f32 (128 KiB) | sQ 32x64 bf16 | sKV 64x64 bf16
// Phase 1: scores (q.k^T * 0.0625 = X*(alpha-1)/sqrt(Dh)) via WMMA -> sS
//          (K tiles DMA'd via TDM tensor_load_to_lds + s_wait_tensorcnt)
// Phase 2: entmax-1.5 bisection, 50 iters, row data register-resident;
//          writes attn f32 -> d_out, attn bf16 aliased over sS rows.
// Phase 3: attn @ v via WMMA -> oh bf16 [4096][1024]
#define NQ 32
__global__ __launch_bounds__(256) void attn_entmax_kernel(const ushort_t* __restrict__ qh,
                                                          const ushort_t* __restrict__ kh,
                                                          const ushort_t* __restrict__ vh,
                                                          float* __restrict__ attn_out,
                                                          ushort_t* __restrict__ oh) {
  extern __shared__ unsigned char smem[];
  float*    sS  = (float*)smem;                                  // 131072 B
  ushort_t* sQ  = (ushort_t*)(smem + 131072);                    //   4096 B
  ushort_t* sKV = (ushort_t*)(smem + 131072 + 4096);             //   8192 B
  const unsigned int sKV_lds = 131072u + 4096u;                  // LDS byte offset

  int bh = blockIdx.x;                 // b*16 + h
  int qt = blockIdx.y;                 // query tile (32 rows)
  int b  = bh >> 4, h = bh & 15;
  int tid = threadIdx.x, wave = tid >> 5, lane = tid & 31;
  int rt = wave >> 2, ct = wave & 3;   // 2x4 tile grid per 32x64 panel

  // load q tile 32x64 bf16
  const ushort_t* qb = qh + ((size_t)bh * 1024 + qt * NQ) * 64;
  for (int e = tid; e < NQ * 64; e += 256) sQ[e] = qb[e];

  int b0  = (lane & 16) ? 16 : 0;
  int b1  = (lane & 16) ? 24 : 8;
  int kb2 = (lane & 16) ? 16 : 0;

  // ---- Phase 1: scores ----
  for (int mt = 0; mt < 1024; mt += 64) {
    if (wave == 0) {   // one TDM descriptor stages the whole 64x64 bf16 K tile
      unsigned long long kaddr =
          (unsigned long long)(const void*)(kh + ((size_t)bh * 1024 + mt) * 64);
      tdm_load_2d_b16(sKV_lds, kaddr, /*rows=*/64, /*cols=*/64,
                      /*stride=*/64, /*tensor_rows=*/1024);
      __builtin_amdgcn_s_wait_tensorcnt(0);
    }
    __syncthreads();

    v8f acc = {};
    #pragma unroll
    for (int kk = 0; kk < 64; kk += 32) {
      int row = rt * 16 + (lane & 15);
      v16bf a = cat16(*(const v8bf*)&sQ[row * 64 + kk + b0],
                      *(const v8bf*)&sQ[row * 64 + kk + b1]);
      int m = ct * 16 + (lane & 15);   // k is [m][d] == B^T, contiguous in d
      v16bf bb = cat16(*(const v8bf*)&sKV[m * 64 + kk + kb2],
                       *(const v8bf*)&sKV[m * 64 + kk + kb2 + 8]);
      acc = wmma_bf16(a, bb, acc);
    }
    #pragma unroll
    for (int r = 0; r < 8; ++r) {
      int lrow = rt * 16 + r + ((lane & 16) ? 8 : 0);
      sS[lrow * 1024 + mt + ct * 16 + (lane & 15)] = acc[r] * 0.0625f;
    }
    __syncthreads();
  }

  // ---- Phase 2: entmax bisection (alpha=1.5 -> p = clip(Xa - tau)^2) ----
  for (int g = 0; g < NQ / 8; ++g) {
    int row = g * 8 + wave;
    float xr[32];
    #pragma unroll
    for (int i = 0; i < 32; ++i) xr[i] = sS[row * 1024 + lane + 32 * i];

    float mx = -3.4e38f;
    #pragma unroll
    for (int i = 0; i < 32; ++i) mx = fmaxf(mx, xr[i]);
    mx = wave_max(mx);

    float tau_lo = mx - 1.0f;
    float tau_hi = mx - 0.03125f;      // (1/1024)^(alpha-1)
    float s = 0.f;
    #pragma unroll
    for (int i = 0; i < 32; ++i) { float t = fmaxf(xr[i] - tau_lo, 0.f); s += t * t; }
    float f_lo = wave_sum(s) - 1.0f;
    float dm = tau_hi - tau_lo, tau_m = tau_lo;

    #pragma unroll 1
    for (int it = 0; it < 50; ++it) {
      dm *= 0.5f;
      tau_m = tau_lo + dm;
      float fm = 0.f;
      #pragma unroll
      for (int i = 0; i < 32; ++i) { float t = fmaxf(xr[i] - tau_m, 0.f); fm += t * t; }
      fm = wave_sum(fm) - 1.0f;
      if (fm * f_lo >= 0.f) tau_lo = tau_m;
    }

    float psum = 0.f;
    #pragma unroll
    for (int i = 0; i < 32; ++i) {
      float t = fmaxf(xr[i] - tau_m, 0.f);
      xr[i] = t * t;                   // reuse regs: xr now holds p
      psum += xr[i];
    }
    float inv = 1.0f / wave_sum(psum);

    float*    arow = attn_out + ((size_t)bh * 1024 + qt * NQ + row) * 1024;
    ushort_t* prow = (ushort_t*)(smem + (size_t)row * 4096);   // alias sS row
    #pragma unroll
    for (int i = 0; i < 32; ++i) {
      float pi = xr[i] * inv;
      arow[lane + 32 * i] = pi;        // coalesced f32 store (the 256MB output)
      prow[lane + 32 * i] = f2bf(pi);  // bf16 copy for attn@v
    }
  }
  __syncthreads();

  // ---- Phase 3: out = attn @ v  (32 x 64, K = 1024 in steps of 32) ----
  const ushort_t* sP = (const ushort_t*)smem;   // rows stride 2048 ushorts
  v8f oacc = {};
  for (int mt = 0; mt < 1024; mt += 32) {
    const ushort_t* vb = vh + ((size_t)bh * 1024 + mt) * 64;
    for (int e = tid; e < 64 * 32; e += 256) {  // stage v^T: [d][m]
      int m = e >> 6, d = e & 63;
      sKV[d * 32 + m] = vb[e];
    }
    __syncthreads();

    int row = rt * 16 + (lane & 15);
    v16bf a = cat16(*(const v8bf*)&sP[row * 2048 + mt + b0],
                    *(const v8bf*)&sP[row * 2048 + mt + b1]);
    int d = ct * 16 + (lane & 15);
    v16bf bb = cat16(*(const v8bf*)&sKV[d * 32 + kb2],
                     *(const v8bf*)&sKV[d * 32 + kb2 + 8]);
    oacc = wmma_bf16(a, bb, oacc);
    __syncthreads();
  }

  #pragma unroll
  for (int r = 0; r < 8; ++r) {
    int lrow = rt * 16 + r + ((lane & 16) ? 8 : 0);
    int grow = b * 1024 + qt * NQ + lrow;
    int gcol = h * 64 + ct * 16 + (lane & 15);
    oh[(size_t)grow * 1024 + gcol] = f2bf(oacc[r]);
  }
}

// ------------------------------- launcher ----------------------------------

extern "C" void kernel_launch(void* const* d_in, const int* in_sizes, int n_in,
                              void* d_out, int out_size, void* d_ws, size_t ws_size,
                              hipStream_t stream) {
  (void)in_sizes; (void)n_in; (void)out_size; (void)ws_size;

  const float* x  = (const float*)d_in[0];
  const float* Wq = (const float*)d_in[1];
  const float* bq = (const float*)d_in[2];
  const float* Wk = (const float*)d_in[3];
  const float* bk = (const float*)d_in[4];
  const float* Wv = (const float*)d_in[5];
  const float* bv = (const float*)d_in[6];
  const float* Wo = (const float*)d_in[7];
  const float* bo = (const float*)d_in[8];

  unsigned char* ws = (unsigned char*)d_ws;
  ushort_t* xb  = (ushort_t*)(ws + 0);
  ushort_t* WqT = (ushort_t*)(ws + (8u  << 20));
  ushort_t* WkT = (ushort_t*)(ws + (10u << 20));
  ushort_t* WvT = (ushort_t*)(ws + (12u << 20));
  ushort_t* WoT = (ushort_t*)(ws + (14u << 20));
  ushort_t* qh  = (ushort_t*)(ws + (16u << 20));
  ushort_t* kh  = (ushort_t*)(ws + (24u << 20));
  ushort_t* vh  = (ushort_t*)(ws + (32u << 20));
  ushort_t* oh  = (ushort_t*)(ws + (40u << 20));

  float* out_f   = (float*)d_out;                 // [4][1024][1024]
  float* attn_f  = out_f + 4ull * 1024 * 1024;    // [4][16][1024][1024]

  // 1) x -> bf16
  cvt_bf16_kernel<<<4194304 / 256, 256, 0, stream>>>(x, xb, 4194304);

  // 2) weight transpose+convert
  dim3 tg(32, 32);
  transpose_bf16_kernel<<<tg, 256, 0, stream>>>(Wq, WqT);
  transpose_bf16_kernel<<<tg, 256, 0, stream>>>(Wk, WkT);
  transpose_bf16_kernel<<<tg, 256, 0, stream>>>(Wv, WvT);
  transpose_bf16_kernel<<<tg, 256, 0, stream>>>(Wo, WoT);

  // 3) q/k/v projections -> head-layout bf16
  dim3 gg(1024 / 128, 4096 / 128);
  gemm_bf16_kernel<<<gg, 256, 0, stream>>>(xb, WqT, bq, qh, 0);
  gemm_bf16_kernel<<<gg, 256, 0, stream>>>(xb, WkT, bk, kh, 0);
  gemm_bf16_kernel<<<gg, 256, 0, stream>>>(xb, WvT, bv, vh, 0);

  // 4) fused scores + entmax(1.5) + attn@v   (143360 B dynamic LDS < 320 KB/WGP)
  dim3 ag(64, 1024 / NQ);
  attn_entmax_kernel<<<ag, 256, 143360, stream>>>(qh, kh, vh, attn_f, oh);

  // 5) output projection -> f32 d_out
  gemm_bf16_kernel<<<gg, 256, 0, stream>>>(oh, WoT, bo, out_f, 1);
}
